// FlowModel_25211458027675
// MI455X (gfx1250) — matrix operand order repaired and checked by reading, not compiled, not run on hardware
//
#include <hip/hip_runtime.h>
#include <math.h>

#define B_    2
#define N_    2048
#define K_    30
#define DN    256
#define DE    128
#define DMSG  640
#define DRED  384              // per-edge GEMM1 K-dim after h_i hoisting
#define NRBF  16
#define BN_   (B_ * N_)
#define EN_   (BN_ * K_)

typedef __bf16 v16bf __attribute__((ext_vector_type(16)));
typedef float  v8f   __attribute__((ext_vector_type(8)));

union FragU { v16bf v; uint4 q[2]; };

__device__ __forceinline__ float softplusf(float x) {
    return (x > 15.f) ? x : log1pf(expf(x));
}

// ---------------------------------------------------------------------------
// Pack f32 weight [layers][Kd][Nd] into WMMA B-fragment tile order (bf16).
// B layout for v_wmma_f32_16x16x32_bf16 (32x16 K x N tile):
//   lanes 0-15 hold K=0..7,16..23 for N=lane ; lanes 16-31 hold K=8..15,24..31
//   per-lane element e (0..15) -> K_local = (lane>=16 ? 8 : 0) + (e & 7) + (e >= 8 ? 16 : 0)
// Packed: ((tile*32 + lane)*16 + e), tile = kt*(Nd/16) + nt
// ---------------------------------------------------------------------------
__global__ void pack_weights_kernel(const float* __restrict__ src,
                                    __bf16* __restrict__ dst,
                                    int Kd, int Nd, int layers) {
    int gid = blockIdx.x * 256 + threadIdx.x;
    int per = Kd * Nd;
    if (gid >= per * layers) return;
    int l = gid / per, r = gid - l * per;
    int k = r / Nd, n = r - k * Nd;
    int kt = k >> 5, kr = k & 31, nt = n >> 4, nr = n & 15;
    int lane = (((kr >> 3) & 1) << 4) + nr;
    int e = (kr & 7) + ((kr >> 4) << 3);
    int NT = Nd >> 4;
    size_t di = (size_t)l * per + (((size_t)(kt * NT + nt) * 32 + lane) * 16 + e);
    dst[di] = (__bf16)src[gid];
}

// ---------------------------------------------------------------------------
// Node init: centroid xc, feats = X - xc, node_h = feats @ W_node + b_node
// ---------------------------------------------------------------------------
__global__ __launch_bounds__(256) void node_init_kernel(
    const float* __restrict__ X, const int* __restrict__ C,
    const float* __restrict__ Wn, const float* __restrict__ bn,
    float* __restrict__ xc, float* __restrict__ h, __bf16* __restrict__ hb) {
    __shared__ float xs[3];
    __shared__ float ft[12];
    int node = blockIdx.x, tid = threadIdx.x;
    const float* xb = X + (size_t)node * 12;
    if (tid < 3) {
        float s = (xb[tid] + xb[3 + tid] + xb[6 + tid] + xb[9 + tid]) * 0.25f;
        xs[tid] = s;
        xc[node * 3 + tid] = s;
    }
    __syncthreads();
    if (tid < 12) ft[tid] = xb[tid] - xs[tid % 3];
    __syncthreads();
    float mi = (C[node] > 0) ? 1.f : 0.f;
    float s = bn[tid];
#pragma unroll
    for (int d = 0; d < 12; ++d) s += ft[d] * Wn[d * DN + tid];
    s *= mi;
    h[(size_t)node * DN + tid] = s;
    hb[(size_t)node * DN + tid] = (__bf16)s;
}

// ---------------------------------------------------------------------------
// kNN: per node, distances to all N nodes, iterative top-30 argmin (stable)
// ---------------------------------------------------------------------------
__global__ __launch_bounds__(256) void knn_kernel(
    const float* __restrict__ xc, const int* __restrict__ C,
    int* __restrict__ eidx, float* __restrict__ dk) {
    __shared__ float dist[N_];
    __shared__ float bd_s[8];
    __shared__ int bi_s[8];
    int node = blockIdx.x, tid = threadIdx.x;
    int b = node / N_;
    int lane = tid & 31, w = tid >> 5;
    float x0 = xc[node * 3 + 0], x1 = xc[node * 3 + 1], x2 = xc[node * 3 + 2];
    float mi = (C[node] > 0) ? 1.f : 0.f;
    for (int j = tid; j < N_; j += 256) {
        const float* pj = xc + ((size_t)b * N_ + j) * 3;
        float d0 = pj[0] - x0, d1 = pj[1] - x1, d2 = pj[2] - x2;
        float d = sqrtf(d0 * d0 + d1 * d1 + d2 * d2 + 1e-8f);
        float mj = (C[b * N_ + j] > 0) ? 1.f : 0.f;
        dist[j] = (mi * mj > 0.f) ? d : 1e6f;
    }
    __syncthreads();
    for (int k = 0; k < K_; ++k) {
        float bd = 3e38f; int bi = N_;
        for (int j = tid; j < N_; j += 256) {
            float d = dist[j];
            if (d < bd) { bd = d; bi = j; }
        }
#pragma unroll
        for (int m = 16; m >= 1; m >>= 1) {
            float od = __shfl_xor(bd, m, 32);
            int   oi = __shfl_xor(bi, m, 32);
            if (od < bd || (od == bd && oi < bi)) { bd = od; bi = oi; }
        }
        if (lane == 0) { bd_s[w] = bd; bi_s[w] = bi; }
        __syncthreads();
        if (tid == 0) {
            float fb = bd_s[0]; int fi = bi_s[0];
#pragma unroll
            for (int x = 1; x < 8; ++x)
                if (bd_s[x] < fb || (bd_s[x] == fb && bi_s[x] < fi)) { fb = bd_s[x]; fi = bi_s[x]; }
            eidx[node * K_ + k] = fi;
            dk[node * K_ + k] = fb;
            dist[fi] = 3e38f;
        }
        __syncthreads();
    }
}

// ---------------------------------------------------------------------------
// Edge init: edge_h = RBF(d) @ W_edge + b_edge, masked
// ---------------------------------------------------------------------------
__global__ __launch_bounds__(128) void edge_init_kernel(
    const float* __restrict__ dk, const int* __restrict__ eidx,
    const int* __restrict__ C, const float* __restrict__ We,
    const float* __restrict__ be, float* __restrict__ eh, __bf16* __restrict__ ehb) {
    int e = blockIdx.x;          // node*K + k
    int node = e / K_;
    int b = node / N_;
    int tid = threadIdx.x;       // output dim 0..127
    float d = dk[e];
    int j = eidx[e];
    float m = (((C[node] > 0) && (C[b * N_ + j] > 0)) ? 1.f : 0.f);
    const float step = 20.f / 15.f;      // linspace(0,20,16)
    const float invsig = 16.f / 20.f;    // 1/sigma, sigma = 1.25
    float s = be[tid];
#pragma unroll
    for (int r = 0; r < NRBF; ++r) {
        float z = (d - step * (float)r) * invsig;
        s += expf(-z * z) * We[r * DE + tid];
    }
    s *= m;
    eh[(size_t)e * DE + tid] = s;
    ehb[(size_t)e * DE + tid] = (__bf16)s;
}

// ---------------------------------------------------------------------------
// Hoisted h_i contribution: P[node][n] = h[node][0:256] @ W1[0:256][n]
// Dense batched GEMM [BN x 256] @ [256 x NDIM], uses packed W1 tiles kt=0..7.
// One workgroup = 32 node-rows, 8 waves cover 2 M-tiles x (NDIM/16) N-tiles.
// ---------------------------------------------------------------------------
template <int NDIM>
__global__ __launch_bounds__(256) void hi_gemm_kernel(
    const __bf16* __restrict__ hb, const __bf16* __restrict__ pW1,
    float* __restrict__ P) {
    constexpr int NT = NDIM / 16;
    constexpr int TPW = (2 * NT) / 8;   // tiles per wave (4 for 256, 2 for 128)
    __shared__ alignas(16) __bf16 Ab[32 * DN];   // 16 KB
    int blk = blockIdx.x;               // 32 nodes
    int tid = threadIdx.x, lane = tid & 31, w = tid >> 5;
    {
        const uint4* hq = (const uint4*)(hb + (size_t)blk * 32 * DN);
        uint4* aq = (uint4*)Ab;
        for (int c = tid; c < 32 * 32; c += 256) aq[c] = hq[c];
    }
    __syncthreads();

    const v8f vz = {0.f, 0.f, 0.f, 0.f, 0.f, 0.f, 0.f, 0.f};
    int mt = (w * TPW) / NT;
    int nbase = (w * TPW) % NT;
    int arow = mt * 16 + (lane & 15);
    int khalf = lane >> 4;

    v8f acc[TPW];
#pragma unroll
    for (int j = 0; j < TPW; ++j) acc[j] = vz;
    const __bf16* Arow = Ab + arow * DN;
    for (int kt = 0; kt < 8; ++kt) {
        FragU fa;
        fa.q[0] = *(const uint4*)(Arow + kt * 32 + khalf * 8);
        fa.q[1] = *(const uint4*)(Arow + kt * 32 + 16 + khalf * 8);
#pragma unroll
        for (int j = 0; j < TPW; ++j) {
            int tile = kt * NT + (nbase + j);
            FragU fb;
            const uint4* bq = (const uint4*)pW1 + (size_t)tile * 64 + lane * 2;
            fb.q[0] = bq[0]; fb.q[1] = bq[1];
            acc[j] = __builtin_amdgcn_wmma_f32_16x16x32_bf16(
                false, fa.v, false, fb.v, (short)0, acc[j], false, false);
        }
    }
#pragma unroll
    for (int j = 0; j < TPW; ++j) {
        int n = (nbase + j) * 16 + (lane & 15);
#pragma unroll
        for (int r = 0; r < 8; ++r) {
            int row = mt * 16 + khalf * 8 + r;
            P[((size_t)blk * 32 + row) * NDIM + n] = acc[j][r];
        }
    }
}

// ---------------------------------------------------------------------------
// Node message layer: one workgroup per node.
//  A[32x384] = [h_j | e]; acc init = P_hi; H = softplus(A@W1[256:]+P+b1);
//  O = H@W2+b2; agg = masked mean over K; node_h' = LN(node_h + agg) * mask
// ---------------------------------------------------------------------------
__global__ __launch_bounds__(256) void node_msg_kernel(
    const __bf16* __restrict__ hb_in, const float* __restrict__ h_in,
    const __bf16* __restrict__ ehb, const int* __restrict__ eidx,
    const int* __restrict__ C, const float* __restrict__ Phi,
    const __bf16* __restrict__ pW1, const __bf16* __restrict__ pW2,
    const float* __restrict__ b1, const float* __restrict__ b2,
    float* __restrict__ h_out, __bf16* __restrict__ hb_out) {
    __shared__ alignas(16) char smem[49472];
    __bf16* Ab = (__bf16*)smem;                 // 32x384 bf16 (24576 B)
    float*  Ob = (float*)smem;                  // 32x256 f32 (32768 B) overlays Ab
    __bf16* Hb = (__bf16*)(smem + 32768);       // 32x256 bf16 (16384 B)
    int*    jrow = (int*)(smem + 49152);
    float*  mrow = (float*)(smem + 49280);
    float*  redS = (float*)(smem + 49408);
    float*  redQ = (float*)(smem + 49440);

    int node = blockIdx.x, tid = threadIdx.x;
    int lane = tid & 31, w = tid >> 5;
    int b = node / N_;
    float mi = (C[node] > 0) ? 1.f : 0.f;

    if (tid < 32) {
        if (tid < K_) {
            int j = eidx[node * K_ + tid];
            jrow[tid] = j;
            mrow[tid] = mi * ((C[b * N_ + j] > 0) ? 1.f : 0.f);
        } else { jrow[tid] = 0; mrow[tid] = 0.f; }
    }
    __syncthreads();

    // ---- stage 1: gather A = [h_j | e] into LDS (uint4 = 8 bf16) ----
    {
        const uint4* hq = (const uint4*)hb_in;   // 32 uint4 / node row
        const uint4* eq = (const uint4*)ehb;     // 16 uint4 / edge row
        uint4* aq = (uint4*)Ab;                  // 48 uint4 / A row
        for (int c = tid; c < 32 * 48; c += 256) {
            int r = c / 48, cc = c - r * 48;
            uint4 v;
            if (r >= K_)      v = make_uint4(0, 0, 0, 0);
            else if (cc < 32) v = hq[((size_t)b * N_ + jrow[r]) * 32 + cc];
            else              v = eq[((size_t)node * K_ + r) * 16 + (cc - 32)];
            aq[c] = v;
        }
    }
    __syncthreads();

    int mt = w >> 2;            // M tile 0..1
    int ng = w & 3;             // N tile group: nt = ng*4 + j
    int arow = mt * 16 + (lane & 15);
    int khalf = lane >> 4;

    // ---- GEMM1: [32x384]@W1[256:640] with acc seeded by hoisted h_i part ----
    v8f acc[4];
#pragma unroll
    for (int j = 0; j < 4; ++j) {
        float p = Phi[(size_t)node * DN + (ng * 4 + j) * 16 + (lane & 15)];
        acc[j] = (v8f){p, p, p, p, p, p, p, p};
    }
    {
        const __bf16* Arow = Ab + arow * DRED;
        for (int kt = 0; kt < 12; ++kt) {
            FragU fa;
            fa.q[0] = *(const uint4*)(Arow + kt * 32 + khalf * 8);
            fa.q[1] = *(const uint4*)(Arow + kt * 32 + 16 + khalf * 8);
#pragma unroll
            for (int j = 0; j < 4; ++j) {
                int tile = (kt + 8) * 16 + (ng * 4 + j);
                FragU fb;
                const uint4* bq = (const uint4*)pW1 + (size_t)tile * 64 + lane * 2;
                fb.q[0] = bq[0]; fb.q[1] = bq[1];
                acc[j] = __builtin_amdgcn_wmma_f32_16x16x32_bf16(
                    false, fa.v, false, fb.v, (short)0, acc[j], false, false);
            }
        }
    }
    // bias + softplus -> Hb (bf16)
#pragma unroll
    for (int j = 0; j < 4; ++j) {
        int n = (ng * 4 + j) * 16 + (lane & 15);
        float bias = b1[n];
#pragma unroll
        for (int r = 0; r < 8; ++r) {
            int row = mt * 16 + khalf * 8 + r;
            Hb[row * DN + n] = (__bf16)softplusf(acc[j][r] + bias);
        }
    }
    __syncthreads();

    // ---- GEMM2: [32x256]@[256x256] ----
    const v8f vz = {0.f, 0.f, 0.f, 0.f, 0.f, 0.f, 0.f, 0.f};
#pragma unroll
    for (int j = 0; j < 4; ++j) acc[j] = vz;
    {
        const __bf16* Hrow = Hb + arow * DN;
        for (int kt = 0; kt < 8; ++kt) {
            FragU fa;
            fa.q[0] = *(const uint4*)(Hrow + kt * 32 + khalf * 8);
            fa.q[1] = *(const uint4*)(Hrow + kt * 32 + 16 + khalf * 8);
#pragma unroll
            for (int j = 0; j < 4; ++j) {
                int tile = kt * 16 + (ng * 4 + j);
                FragU fb;
                const uint4* bq = (const uint4*)pW2 + (size_t)tile * 64 + lane * 2;
                fb.q[0] = bq[0]; fb.q[1] = bq[1];
                acc[j] = __builtin_amdgcn_wmma_f32_16x16x32_bf16(
                    false, fa.v, false, fb.v, (short)0, acc[j], false, false);
            }
        }
    }
#pragma unroll
    for (int j = 0; j < 4; ++j) {
        int n = (ng * 4 + j) * 16 + (lane & 15);
        float bias = b2[n];
#pragma unroll
        for (int r = 0; r < 8; ++r) {
            int row = mt * 16 + khalf * 8 + r;
            Ob[row * DN + n] = acc[j][r] + bias;
        }
    }
    __syncthreads();

    // ---- aggregation + residual + LayerNorm (thread = output dim) ----
    float denom = 1e-8f;
    float s = 0.f;
    for (int k = 0; k < K_; ++k) {
        float mk = mrow[k];
        denom += mk;
        s += mk * Ob[k * DN + tid];
    }
    float val = h_in[(size_t)node * DN + tid] + s / denom;
    float ss = val, qq = val * val;
#pragma unroll
    for (int m = 16; m >= 1; m >>= 1) {
        ss += __shfl_xor(ss, m, 32);
        qq += __shfl_xor(qq, m, 32);
    }
    if (lane == 0) { redS[w] = ss; redQ[w] = qq; }
    __syncthreads();
    float mean = 0.f, msq = 0.f;
#pragma unroll
    for (int x = 0; x < 8; ++x) { mean += redS[x]; msq += redQ[x]; }
    mean *= (1.f / 256.f); msq *= (1.f / 256.f);
    float var = msq - mean * mean;
    float o = (val - mean) * rsqrtf(var + 1e-5f) * mi;
    h_out[(size_t)node * DN + tid] = o;
    hb_out[(size_t)node * DN + tid] = (__bf16)o;
}

// ---------------------------------------------------------------------------
// Edge update layer: one workgroup per node (its 30 edges).
//  A[32x384] = [h_j | e] (updated h); acc init = Pe (hoisted h_i part);
//  H = softplus(...); O = H@We2+be2; edge_h' = LN(edge_h + O) * mask  (in place)
// ---------------------------------------------------------------------------
__global__ __launch_bounds__(256) void edge_upd_kernel(
    const __bf16* __restrict__ hb_in, const int* __restrict__ eidx,
    const int* __restrict__ C, const float* __restrict__ Pe,
    const __bf16* __restrict__ pW1, const __bf16* __restrict__ pW2,
    const float* __restrict__ b1, const float* __restrict__ b2,
    float* __restrict__ eh, __bf16* __restrict__ ehb) {
    __shared__ alignas(16) char smem[33280];
    __bf16* Ab = (__bf16*)smem;                 // 32x384 bf16 (24576 B)
    float*  Ob = (float*)smem;                  // 32x128 f32 (16384 B) overlays Ab
    __bf16* Hb = (__bf16*)(smem + 24576);       // 32x128 bf16 (8192 B)
    int*    jrow = (int*)(smem + 32768);
    float*  mrow = (float*)(smem + 32896);
    float*  muv  = (float*)(smem + 33024);
    float*  rsv  = (float*)(smem + 33152);

    int node = blockIdx.x, tid = threadIdx.x;
    int lane = tid & 31, w = tid >> 5;
    int b = node / N_;
    float mi = (C[node] > 0) ? 1.f : 0.f;

    if (tid < 32) {
        if (tid < K_) {
            int j = eidx[node * K_ + tid];
            jrow[tid] = j;
            mrow[tid] = mi * ((C[b * N_ + j] > 0) ? 1.f : 0.f);
        } else { jrow[tid] = 0; mrow[tid] = 0.f; }
    }
    __syncthreads();

    {
        const uint4* hq = (const uint4*)hb_in;
        const uint4* eq = (const uint4*)ehb;
        uint4* aq = (uint4*)Ab;
        for (int c = tid; c < 32 * 48; c += 256) {
            int r = c / 48, cc = c - r * 48;
            uint4 v;
            if (r >= K_)      v = make_uint4(0, 0, 0, 0);
            else if (cc < 32) v = hq[((size_t)b * N_ + jrow[r]) * 32 + cc];
            else              v = eq[((size_t)node * K_ + r) * 16 + (cc - 32)];
            aq[c] = v;
        }
    }
    __syncthreads();

    int mt = w >> 2;
    int ng = w & 3;               // nt = ng + 4*j, j in {0,1}
    int arow = mt * 16 + (lane & 15);
    int khalf = lane >> 4;

    // ---- GEMM1: [32x384]@We1[256:640] seeded by hoisted h_i part ----
    v8f acc[2];
#pragma unroll
    for (int j = 0; j < 2; ++j) {
        float p = Pe[(size_t)node * DE + (ng + 4 * j) * 16 + (lane & 15)];
        acc[j] = (v8f){p, p, p, p, p, p, p, p};
    }
    {
        const __bf16* Arow = Ab + arow * DRED;
        for (int kt = 0; kt < 12; ++kt) {
            FragU fa;
            fa.q[0] = *(const uint4*)(Arow + kt * 32 + khalf * 8);
            fa.q[1] = *(const uint4*)(Arow + kt * 32 + 16 + khalf * 8);
#pragma unroll
            for (int j = 0; j < 2; ++j) {
                int tile = (kt + 8) * 8 + (ng + 4 * j);
                FragU fb;
                const uint4* bq = (const uint4*)pW1 + (size_t)tile * 64 + lane * 2;
                fb.q[0] = bq[0]; fb.q[1] = bq[1];
                acc[j] = __builtin_amdgcn_wmma_f32_16x16x32_bf16(
                    false, fa.v, false, fb.v, (short)0, acc[j], false, false);
            }
        }
    }
#pragma unroll
    for (int j = 0; j < 2; ++j) {
        int n = (ng + 4 * j) * 16 + (lane & 15);
        float bias = b1[n];
#pragma unroll
        for (int r = 0; r < 8; ++r) {
            int row = mt * 16 + khalf * 8 + r;
            Hb[row * DE + n] = (__bf16)softplusf(acc[j][r] + bias);
        }
    }
    __syncthreads();

    // ---- GEMM2: [32x128]@[128x128] ----
    const v8f vz = {0.f, 0.f, 0.f, 0.f, 0.f, 0.f, 0.f, 0.f};
    acc[0] = vz; acc[1] = vz;
    {
        const __bf16* Hrow = Hb + arow * DE;
        for (int kt = 0; kt < 4; ++kt) {
            FragU fa;
            fa.q[0] = *(const uint4*)(Hrow + kt * 32 + khalf * 8);
            fa.q[1] = *(const uint4*)(Hrow + kt * 32 + 16 + khalf * 8);
#pragma unroll
            for (int j = 0; j < 2; ++j) {
                int tile = kt * 8 + (ng + 4 * j);
                FragU fb;
                const uint4* bq = (const uint4*)pW2 + (size_t)tile * 64 + lane * 2;
                fb.q[0] = bq[0]; fb.q[1] = bq[1];
                acc[j] = __builtin_amdgcn_wmma_f32_16x16x32_bf16(
                    false, fa.v, false, fb.v, (short)0, acc[j], false, false);
            }
        }
    }
#pragma unroll
    for (int j = 0; j < 2; ++j) {
        int n = (ng + 4 * j) * 16 + (lane & 15);
        float bias = b2[n];
#pragma unroll
        for (int r = 0; r < 8; ++r) {
            int row = mt * 16 + khalf * 8 + r;
            Ob[row * DE + n] = acc[j][r] + bias;
        }
    }
    __syncthreads();

    // ---- residual add (old edge_h) into Ob, rows < K ----
    {
        float4* of = (float4*)Ob;
        const float4* ef = (const float4*)(eh + (size_t)node * K_ * DE);
        for (int c = tid; c < K_ * (DE / 4); c += 256) {   // 30 rows x 32 float4
            float4 e4 = ef[c];
            float4 o4 = of[c];
            o4.x += e4.x; o4.y += e4.y; o4.z += e4.z; o4.w += e4.w;
            of[c] = o4;
        }
    }
    __syncthreads();

    // ---- per-row LN stats ----
    if (tid < K_) {
        float m = 0.f, q = 0.f;
        for (int c = 0; c < DE; ++c) { float v = Ob[tid * DE + c]; m += v; q += v * v; }
        m *= (1.f / (float)DE); q *= (1.f / (float)DE);
        muv[tid] = m;
        rsv[tid] = rsqrtf(q - m * m + 1e-5f);
    }
    __syncthreads();

    // ---- normalize + mask, write back (f32 + bf16) ----
    {
        size_t base = (size_t)node * K_ * DE;
        for (int c = tid; c < K_ * DE; c += 256) {
            int r = c >> 7;  // /128
            float o = (Ob[c] - muv[r]) * rsv[r] * mrow[r];
            eh[base + c] = o;
            ehb[base + c] = (__bf16)o;
        }
    }
}

// ---------------------------------------------------------------------------
// Finalize: d_out = [node_h | edge_h | edge_idx(float) | mask_i | mask_ij]
// ---------------------------------------------------------------------------
__global__ void finalize_kernel(const float* __restrict__ h,
                                const float* __restrict__ eh,
                                const int* __restrict__ eidx,
                                const int* __restrict__ C,
                                float* __restrict__ out) {
    const long long NH = (long long)BN_ * DN;
    const long long NE = (long long)EN_ * DE;
    const long long NI = EN_;
    const long long NM = BN_;
    const long long total = NH + NE + NI + NM + NI;
    for (long long i = (long long)blockIdx.x * 256 + threadIdx.x; i < total;
         i += (long long)gridDim.x * 256) {
        float v; long long r = i;
        if (r < NH) v = h[r];
        else {
            r -= NH;
            if (r < NE) v = eh[r];
            else {
                r -= NE;
                if (r < NI) v = (float)eidx[r];
                else {
                    r -= NI;
                    if (r < NM) v = (C[r] > 0) ? 1.f : 0.f;
                    else {
                        r -= NM;
                        int node = (int)(r / K_);
                        int b = node / N_;
                        int j = eidx[r];
                        v = ((C[node] > 0) && (C[b * N_ + j] > 0)) ? 1.f : 0.f;
                    }
                }
            }
        }
        out[i] = v;
    }
}

// ---------------------------------------------------------------------------
extern "C" void kernel_launch(void* const* d_in, const int* in_sizes, int n_in,
                              void* d_out, int out_size, void* d_ws, size_t ws_size,
                              hipStream_t stream) {
    (void)in_sizes; (void)n_in; (void)out_size; (void)ws_size;
    const float* X      = (const float*)d_in[0];
    const int*   C      = (const int*)d_in[1];
    const float* W_node = (const float*)d_in[2];
    const float* b_node = (const float*)d_in[3];
    const float* W_edge = (const float*)d_in[4];
    const float* b_edge = (const float*)d_in[5];
    const float* Wn1    = (const float*)d_in[6];
    const float* bn1    = (const float*)d_in[7];
    const float* Wn2    = (const float*)d_in[8];
    const float* bn2    = (const float*)d_in[9];
    const float* We1    = (const float*)d_in[10];
    const float* be1    = (const float*)d_in[11];
    const float* We2    = (const float*)d_in[12];
    const float* be2    = (const float*)d_in[13];

    char* ws = (char*)d_ws;
    size_t off = 0;
    auto alloc = [&](size_t bytes) -> char* {
        char* p = ws + off;
        off += (bytes + 255) & ~(size_t)255;
        return p;
    };
    float*  xc   = (float*)alloc((size_t)BN_ * 3 * 4);
    int*    eidx = (int*)alloc((size_t)BN_ * K_ * 4);
    float*  dk   = (float*)alloc((size_t)BN_ * K_ * 4);
    float*  hA   = (float*)alloc((size_t)BN_ * DN * 4);
    float*  hB   = (float*)alloc((size_t)BN_ * DN * 4);
    __bf16* hbA  = (__bf16*)alloc((size_t)BN_ * DN * 2);
    __bf16* hbB  = (__bf16*)alloc((size_t)BN_ * DN * 2);
    float*  eh   = (float*)alloc((size_t)EN_ * DE * 4);
    __bf16* ehb  = (__bf16*)alloc((size_t)EN_ * DE * 2);
    float*  Pn   = (float*)alloc((size_t)BN_ * DN * 4);
    float*  Pe   = (float*)alloc((size_t)BN_ * DE * 4);
    __bf16* pWn1 = (__bf16*)alloc((size_t)3 * DMSG * DN * 2);
    __bf16* pWn2 = (__bf16*)alloc((size_t)3 * DN * DN * 2);
    __bf16* pWe1 = (__bf16*)alloc((size_t)3 * DMSG * DE * 2);
    __bf16* pWe2 = (__bf16*)alloc((size_t)3 * DE * DE * 2);

    // Pack weights into WMMA tile order (bf16)
    {
        int t;
        t = 3 * DMSG * DN; pack_weights_kernel<<<(t + 255) / 256, 256, 0, stream>>>(Wn1, pWn1, DMSG, DN, 3);
        t = 3 * DN * DN;   pack_weights_kernel<<<(t + 255) / 256, 256, 0, stream>>>(Wn2, pWn2, DN, DN, 3);
        t = 3 * DMSG * DE; pack_weights_kernel<<<(t + 255) / 256, 256, 0, stream>>>(We1, pWe1, DMSG, DE, 3);
        t = 3 * DE * DE;   pack_weights_kernel<<<(t + 255) / 256, 256, 0, stream>>>(We2, pWe2, DE, DE, 3);
    }

    node_init_kernel<<<BN_, 256, 0, stream>>>(X, C, W_node, b_node, xc, hA, hbA);
    knn_kernel<<<BN_, 256, 0, stream>>>(xc, C, eidx, dk);
    edge_init_kernel<<<EN_, 128, 0, stream>>>(dk, eidx, C, W_edge, b_edge, eh, ehb);

    float*  hc = hA;  float*  hn = hB;
    __bf16* hbc = hbA; __bf16* hbn = hbB;
    for (int l = 0; l < 3; ++l) {
        hi_gemm_kernel<DN><<<BN_ / 32, 256, 0, stream>>>(
            hbc, pWn1 + (size_t)l * DMSG * DN, Pn);
        node_msg_kernel<<<BN_, 256, 0, stream>>>(
            hbc, hc, ehb, eidx, C, Pn,
            pWn1 + (size_t)l * DMSG * DN, pWn2 + (size_t)l * DN * DN,
            bn1 + l * DN, bn2 + l * DN, hn, hbn);
        { float* t = hc; hc = hn; hn = t; }
        { __bf16* t = hbc; hbc = hbn; hbn = t; }
        hi_gemm_kernel<DE><<<BN_ / 32, 256, 0, stream>>>(
            hbc, pWe1 + (size_t)l * DMSG * DE, Pe);
        edge_upd_kernel<<<BN_, 256, 0, stream>>>(
            hbc, eidx, C, Pe,
            pWe1 + (size_t)l * DMSG * DE, pWe2 + (size_t)l * DE * DE,
            be1 + l * DE, be2 + l * DE, eh, ehb);
    }

    finalize_kernel<<<8192, 256, 0, stream>>>(hc, eh, eidx, C, (float*)d_out);
}
